// LSTM_9088150798388
// MI455X (gfx1250) — matrix operand order repaired and checked by reading, compiled, not run on hardware
//
#include <hip/hip_runtime.h>

// ---------------------------------------------------------------------------
// CDNA5 WMMA types / fragment helpers (wave32, 16x16x32 f16 -> f32)
// ---------------------------------------------------------------------------
typedef __attribute__((ext_vector_type(16))) _Float16 v16h;
typedef __attribute__((ext_vector_type(8)))  float    v8f;

union Frag16 { v16h v; unsigned int u[8]; };

// A-matrix 16x32 f16 fragment from a row-major [rows][stride] buffer.
// Layout (ISA 7.12.2): lane L: m=L&15; vgpr r holds k = (r/4)*16 + (L>>4)*8 + (r&3)*2
__device__ __forceinline__ v16h load_a16(const _Float16* base, int stride,
                                         int row0, int k0, int lane) {
  Frag16 f;
  const int m  = lane & 15;
  const int kh = (lane >> 4) << 3;
  const unsigned int* p =
      reinterpret_cast<const unsigned int*>(base + (size_t)(row0 + m) * stride);
#pragma unroll
  for (int r = 0; r < 8; ++r) {
    const int k = ((r >> 2) << 4) + kh + ((r & 3) << 1);
    f.u[r] = p[(k0 + k) >> 1];
  }
  return f.v;
}

// B-matrix 32x16 f16 fragment. Source stored "n-major": W[n][k], row length K halves.
// Layout (same as A): vgpr r holds k = (r/4)*16 + (L>>4)*8 + (r&3)*2, +1.
__device__ __forceinline__ v16h load_b16(const _Float16* W, int K,
                                         int n0, int k0, int lane) {
  Frag16 f;
  const int n  = n0 + (lane & 15);
  const int kg = (lane >> 4) << 3;
  const unsigned int* p = reinterpret_cast<const unsigned int*>(W + (size_t)n * K);
#pragma unroll
  for (int r = 0; r < 8; ++r) f.u[r] = p[(k0 + ((r >> 2) << 4) + kg + ((r & 3) << 1)) >> 1];
  return f.v;
}

__device__ __forceinline__ v8f wmma16(v16h a, v16h b, v8f c) {
  return __builtin_amdgcn_wmma_f32_16x16x32_f16(false, a, false, b,
                                                (short)0, c, false, false);
}

__device__ __forceinline__ v8f splat8(float x) {
  v8f a;
#pragma unroll
  for (int e = 0; e < 8; ++e) a[e] = x;
  return a;
}

__device__ __forceinline__ float sigmoidf(float x) { return 1.0f / (1.0f + expf(-x)); }

// ---------------------------------------------------------------------------
// Problem constants
// ---------------------------------------------------------------------------
#define Bsz   4096
#define Tsz   60
#define Fsz   64
#define Hsz   128
#define G4H   512          // 4*H
#define K1    192          // F + H  (layer-1 concat K)
#define K2    256          // H + H  (layer-2 concat K)
#define FC1K  7680         // T*H
#define FC1N  512
#define FC2N  60

// ---------------------------------------------------------------------------
// prep: fuse biases, convert/concat weights to f16
// ---------------------------------------------------------------------------
__global__ void prep_kernel(const float* __restrict__ Wih1, const float* __restrict__ Whh1,
                            const float* __restrict__ bih1, const float* __restrict__ bhh1,
                            const float* __restrict__ Wih2, const float* __restrict__ Whh2,
                            const float* __restrict__ bih2, const float* __restrict__ bhh2,
                            const float* __restrict__ Wfc1, const float* __restrict__ bfc1,
                            const float* __restrict__ Wfc2, const float* __restrict__ bfc2,
                            _Float16* __restrict__ wc1, _Float16* __restrict__ wc2,
                            float* __restrict__ bs1, float* __restrict__ bs2,
                            _Float16* __restrict__ wf1, float* __restrict__ bf1o,
                            _Float16* __restrict__ wf2, float* __restrict__ bf2o) {
  const int i = blockIdx.x * blockDim.x + threadIdx.x;
  if (i < G4H * FC1K) wf1[i] = (_Float16)Wfc1[i];
  if (i < G4H * K1) {
    const int n = i / K1, k = i % K1;
    wc1[i] = (_Float16)((k < Fsz) ? Wih1[n * Fsz + k] : Whh1[n * Hsz + (k - Fsz)]);
  }
  if (i < G4H * K2) {
    const int n = i / K2, k = i % K2;
    wc2[i] = (_Float16)((k < Hsz) ? Wih2[n * Hsz + k] : Whh2[n * Hsz + (k - Hsz)]);
  }
  if (i < 64 * FC1N) {
    const int n = i / FC1N, k = i % FC1N;
    wf2[i] = (n < FC2N) ? (_Float16)Wfc2[n * FC1N + k] : (_Float16)0.0f;
  }
  if (i < G4H) {
    bs1[i] = bih1[i] + bhh1[i];
    bs2[i] = bih2[i] + bhh2[i];
    if (i < FC1N) bf1o[i] = bfc1[i];
  }
  if (i < FC2N) bf2o[i] = bfc2[i];
}

// ---------------------------------------------------------------------------
// Fused 2-layer LSTM. One WG = 64 batch rows, 8 waves.
// Wave w owns output columns j in [16w, 16w+16) and ALL FOUR gate tiles for
// that j, making the cell update wave-local (c held in registers).
// K-loops are deliberately NOT unrolled: B fragments must be loaded
// just-in-time to keep the live set small (no scratch spills in the
// recurrence-critical path).
// ---------------------------------------------------------------------------
__global__ void __launch_bounds__(256) lstm2_fused(
    const float*    __restrict__ X,     // [B][T][F] fp32
    const _Float16* __restrict__ Wc1,   // [512][192] f16 ([Wih1|Whh1])
    const _Float16* __restrict__ Wc2,   // [512][256] f16 ([Wih2|Whh2])
    const float*    __restrict__ bias1, // [512]
    const float*    __restrict__ bias2, // [512]
    _Float16*       __restrict__ H2g)   // [B][T][H] f16  (== [B][7680])
{
  __shared__ alignas(16) _Float16 sXt[64 * Fsz];   //  8 KB : x_t tile
  __shared__ alignas(16) _Float16 sH1[64 * Hsz];   // 16 KB : layer-1 hidden
  __shared__ alignas(16) _Float16 sH2[64 * Hsz];   // 16 KB : layer-2 hidden

  const int tid  = threadIdx.x;
  const int lane = tid & 31;
  const int wave = tid >> 5;
  const int b0   = blockIdx.x * 64;
  const int j0   = wave * 16;          // this wave's hidden-column slice
  const int nlo  = lane & 15;
  const int mhi  = (lane >> 4) << 3;

  for (int i = tid; i < 64 * Hsz; i += 256) {
    sH1[i] = (_Float16)0.0f;
    sH2[i] = (_Float16)0.0f;
  }

  float c1[4][8], c2[4][8];
#pragma unroll
  for (int mt = 0; mt < 4; ++mt)
#pragma unroll
    for (int v = 0; v < 8; ++v) { c1[mt][v] = 0.0f; c2[mt][v] = 0.0f; }

  float bia1[4], bia2[4];
#pragma unroll
  for (int g = 0; g < 4; ++g) {
    bia1[g] = bias1[g * Hsz + j0 + nlo];
    bia2[g] = bias2[g * Hsz + j0 + nlo];
  }
  __syncthreads();

#pragma unroll 1
  for (int t = 0; t < Tsz; ++t) {
    // ---- stage x_t (fp32 -> f16) into LDS, coalesced float4 loads ----
    {
      const int row = tid >> 2;
      const int seg = (tid & 3) * 16;
      const float4* src = reinterpret_cast<const float4*>(
          X + ((size_t)(b0 + row) * Tsz + t) * Fsz + seg);
#pragma unroll
      for (int q = 0; q < 4; ++q) {
        const float4 xv = src[q];
        const int o = row * Fsz + seg + q * 4;
        sXt[o + 0] = (_Float16)xv.x;  sXt[o + 1] = (_Float16)xv.y;
        sXt[o + 2] = (_Float16)xv.z;  sXt[o + 3] = (_Float16)xv.w;
      }
    }
    __syncthreads();

    // =================== layer 1: [x_t|h1] (64x192) x Wc1^T ===================
    float h1v[4][8];
#pragma unroll 1
    for (int mp = 0; mp < 2; ++mp) {       // M-tile pairs -> bounded reg pressure
      v8f acc[2][4];
#pragma unroll
      for (int i = 0; i < 2; ++i)
#pragma unroll
        for (int g = 0; g < 4; ++g) acc[i][g] = splat8(bia1[g]);

      // K-chunks 0..1: A from x_t tile
#pragma unroll 1
      for (int kc = 0; kc < Fsz / 32; ++kc) {
        const int k0 = kc * 32;
        v16h bf[4];
#pragma unroll
        for (int g = 0; g < 4; ++g) bf[g] = load_b16(Wc1, K1, g * Hsz + j0, k0, lane);
#pragma unroll
        for (int i = 0; i < 2; ++i) {
          const v16h af = load_a16(sXt, Fsz, (mp * 2 + i) * 16, k0, lane);
#pragma unroll
          for (int g = 0; g < 4; ++g) acc[i][g] = wmma16(af, bf[g], acc[i][g]);
        }
      }
      // K-chunks 2..5: A from h1 (previous step)
#pragma unroll 1
      for (int kc = 0; kc < Hsz / 32; ++kc) {
        const int k0 = kc * 32;
        v16h bf[4];
#pragma unroll
        for (int g = 0; g < 4; ++g)
          bf[g] = load_b16(Wc1, K1, g * Hsz + j0, Fsz + k0, lane);
#pragma unroll
        for (int i = 0; i < 2; ++i) {
          const v16h af = load_a16(sH1, Hsz, (mp * 2 + i) * 16, k0, lane);
#pragma unroll
          for (int g = 0; g < 4; ++g) acc[i][g] = wmma16(af, bf[g], acc[i][g]);
        }
      }
#pragma unroll
      for (int i = 0; i < 2; ++i) {
        const int mt = mp * 2 + i;
#pragma unroll
        for (int v = 0; v < 8; ++v) {
          const float iv = sigmoidf(acc[i][0][v]);
          const float fv = sigmoidf(acc[i][1][v]);
          const float gv = tanhf(acc[i][2][v]);
          const float ov = sigmoidf(acc[i][3][v]);
          const float c  = fv * c1[mt][v] + iv * gv;
          c1[mt][v] = c;
          h1v[mt][v] = ov * tanhf(c);
        }
      }
    }
    __syncthreads();                       // everyone done reading old sH1
#pragma unroll
    for (int mt = 0; mt < 4; ++mt)
#pragma unroll
      for (int v = 0; v < 8; ++v)
        sH1[(mt * 16 + mhi + v) * Hsz + j0 + nlo] = (_Float16)h1v[mt][v];
    __syncthreads();

    // =================== layer 2: [h1_t|h2] (64x256) x Wc2^T ==================
    float h2v[4][8];
#pragma unroll 1
    for (int mp = 0; mp < 2; ++mp) {
      v8f acc[2][4];
#pragma unroll
      for (int i = 0; i < 2; ++i)
#pragma unroll
        for (int g = 0; g < 4; ++g) acc[i][g] = splat8(bia2[g]);

      // K-chunks 0..3: A from h1_t (just written)
#pragma unroll 1
      for (int kc = 0; kc < Hsz / 32; ++kc) {
        const int k0 = kc * 32;
        v16h bf[4];
#pragma unroll
        for (int g = 0; g < 4; ++g) bf[g] = load_b16(Wc2, K2, g * Hsz + j0, k0, lane);
#pragma unroll
        for (int i = 0; i < 2; ++i) {
          const v16h af = load_a16(sH1, Hsz, (mp * 2 + i) * 16, k0, lane);
#pragma unroll
          for (int g = 0; g < 4; ++g) acc[i][g] = wmma16(af, bf[g], acc[i][g]);
        }
      }
      // K-chunks 4..7: A from h2 (previous step)
#pragma unroll 1
      for (int kc = 0; kc < Hsz / 32; ++kc) {
        const int k0 = kc * 32;
        v16h bf[4];
#pragma unroll
        for (int g = 0; g < 4; ++g)
          bf[g] = load_b16(Wc2, K2, g * Hsz + j0, Hsz + k0, lane);
#pragma unroll
        for (int i = 0; i < 2; ++i) {
          const v16h af = load_a16(sH2, Hsz, (mp * 2 + i) * 16, k0, lane);
#pragma unroll
          for (int g = 0; g < 4; ++g) acc[i][g] = wmma16(af, bf[g], acc[i][g]);
        }
      }
#pragma unroll
      for (int i = 0; i < 2; ++i) {
        const int mt = mp * 2 + i;
#pragma unroll
        for (int v = 0; v < 8; ++v) {
          const float iv = sigmoidf(acc[i][0][v]);
          const float fv = sigmoidf(acc[i][1][v]);
          const float gv = tanhf(acc[i][2][v]);
          const float ov = sigmoidf(acc[i][3][v]);
          const float c  = fv * c2[mt][v] + iv * gv;
          c2[mt][v] = c;
          h2v[mt][v] = ov * tanhf(c);
        }
      }
    }
    __syncthreads();                       // everyone done reading old sH2
#pragma unroll
    for (int mt = 0; mt < 4; ++mt)
#pragma unroll
      for (int v = 0; v < 8; ++v) {
        const int m = mt * 16 + mhi + v;
        const int j = j0 + nlo;
        const _Float16 hv = (_Float16)h2v[mt][v];
        sH2[m * Hsz + j] = hv;
        H2g[((size_t)(b0 + m) * Tsz + t) * Hsz + j] = hv;
      }
    __syncthreads();
  }
}

// ---------------------------------------------------------------------------
// FC1: [4096,7680] f16 x [7680,512] -> f16 [4096,512] (+bias), LDS-staged tiles
// grid (32, 8): 128 rows x 64 cols per WG, wave w = M-tile, 4 N-tiles each.
// ---------------------------------------------------------------------------
__global__ void __launch_bounds__(256) fc1_gemm(
    const _Float16* __restrict__ A,    // [4096][7680]
    const _Float16* __restrict__ W,    // [512][7680] (n-major)
    const float*    __restrict__ bias, // [512]
    _Float16*       __restrict__ Out)  // [4096][512]
{
  __shared__ alignas(16) _Float16 sA[128 * 32];
  __shared__ alignas(16) _Float16 sB[64 * 32];

  const int tid  = threadIdx.x;
  const int lane = tid & 31;
  const int wave = tid >> 5;
  const int m0   = blockIdx.x * 128;
  const int n0   = blockIdx.y * 64;
  const int nlo  = lane & 15;
  const int mhi  = (lane >> 4) << 3;

  v8f acc[4];
#pragma unroll
  for (int nt = 0; nt < 4; ++nt) acc[nt] = splat8(bias[n0 + nt * 16 + nlo]);

#pragma unroll 1
  for (int kc = 0; kc < FC1K / 32; ++kc) {
    const int k0 = kc * 32;
    {   // stage A: 128 rows x 32 halves; 2 threads/row, 2x b128 each
      const int row = tid >> 1;
      const int hp  = (tid & 1) * 16;
      const uint4* src = reinterpret_cast<const uint4*>(
          A + (size_t)(m0 + row) * FC1K + k0 + hp);
      uint4* dst = reinterpret_cast<uint4*>(sA + row * 32 + hp);
      dst[0] = src[0];
      dst[1] = src[1];
    }
    {   // stage B: 64 rows x 32 halves; 4 threads/row, 1x b128 each
      const int row = tid >> 2;
      const int hp  = (tid & 3) * 8;
      *reinterpret_cast<uint4*>(sB + row * 32 + hp) =
          *reinterpret_cast<const uint4*>(W + (size_t)(n0 + row) * FC1K + k0 + hp);
    }
    __syncthreads();
    const v16h af = load_a16(sA, 32, wave * 16, 0, lane);
#pragma unroll
    for (int nt = 0; nt < 4; ++nt) {
      const v16h bf = load_b16(sB, 32, nt * 16, 0, lane);
      acc[nt] = wmma16(af, bf, acc[nt]);
    }
    __syncthreads();
  }
#pragma unroll
  for (int nt = 0; nt < 4; ++nt)
#pragma unroll
    for (int v = 0; v < 8; ++v)
      Out[(size_t)(m0 + wave * 16 + mhi + v) * FC1N + n0 + nt * 16 + nlo] =
          (_Float16)acc[nt][v];
}

// ---------------------------------------------------------------------------
// FC2: [4096,512] f16 x [512,64(pad of 60)] -> fp32 d_out [4096,60] (+bias)
// ---------------------------------------------------------------------------
__global__ void __launch_bounds__(256) fc2_gemm(
    const _Float16* __restrict__ A,    // [4096][512]
    const _Float16* __restrict__ W,    // [64][512] rows >= 60 are zero
    const float*    __restrict__ bias, // [60]
    float*          __restrict__ Out)  // [4096][60]
{
  const int tid  = threadIdx.x;
  const int lane = tid & 31;
  const int wave = tid >> 5;
  const int m0   = blockIdx.x * 128;
  const int nlo  = lane & 15;
  const int mhi  = (lane >> 4) << 3;

  v8f acc[4];
#pragma unroll
  for (int nt = 0; nt < 4; ++nt) {
    const int n = nt * 16 + nlo;
    acc[nt] = splat8(n < FC2N ? bias[n] : 0.0f);
  }
#pragma unroll 1
  for (int kc = 0; kc < FC1N / 32; ++kc) {
    const int k0 = kc * 32;
    const v16h af = load_a16(A, FC1N, m0 + wave * 16, k0, lane);
#pragma unroll
    for (int nt = 0; nt < 4; ++nt) {
      const v16h bf = load_b16(W, FC1N, nt * 16, k0, lane);
      acc[nt] = wmma16(af, bf, acc[nt]);
    }
  }
#pragma unroll
  for (int nt = 0; nt < 4; ++nt) {
    const int n = nt * 16 + nlo;
    if (n < FC2N) {
#pragma unroll
      for (int v = 0; v < 8; ++v) {
        const int m = m0 + wave * 16 + mhi + v;
        Out[(size_t)m * FC2N + n] = acc[nt][v];
      }
    }
  }
}

// ---------------------------------------------------------------------------
// Host launcher
// ---------------------------------------------------------------------------
extern "C" void kernel_launch(void* const* d_in, const int* in_sizes, int n_in,
                              void* d_out, int out_size, void* d_ws, size_t ws_size,
                              hipStream_t stream) {
  const float* X    = (const float*)d_in[0];
  const float* Wih1 = (const float*)d_in[1];
  const float* Whh1 = (const float*)d_in[2];
  const float* bih1 = (const float*)d_in[3];
  const float* bhh1 = (const float*)d_in[4];
  const float* Wih2 = (const float*)d_in[5];
  const float* Whh2 = (const float*)d_in[6];
  const float* bih2 = (const float*)d_in[7];
  const float* bhh2 = (const float*)d_in[8];
  const float* Wfc1 = (const float*)d_in[9];
  const float* bfc1 = (const float*)d_in[10];
  const float* Wfc2 = (const float*)d_in[11];
  const float* bfc2 = (const float*)d_in[12];

  char* base = (char*)d_ws;
  size_t off = 0;
  auto carve = [&](size_t bytes) -> void* {
    void* p = base + off;
    off = (off + bytes + 255) & ~(size_t)255;
    return p;
  };
  _Float16* wc1  = (_Float16*)carve((size_t)G4H * K1 * 2);         // 196 KB
  _Float16* wc2  = (_Float16*)carve((size_t)G4H * K2 * 2);         // 262 KB
  float*    bs1  = (float*)carve(G4H * 4);
  float*    bs2  = (float*)carve(G4H * 4);
  _Float16* wf1  = (_Float16*)carve((size_t)FC1N * FC1K * 2);      // 7.9 MB
  float*    bf1  = (float*)carve(FC1N * 4);
  _Float16* wf2  = (_Float16*)carve((size_t)64 * FC1N * 2);        // 64 KB
  float*    bf2  = (float*)carve(64 * 4);
  _Float16* h2g  = (_Float16*)carve((size_t)Bsz * FC1K * 2);       // 63 MB
  _Float16* fc1o = (_Float16*)carve((size_t)Bsz * FC1N * 2);       // 4.2 MB

  {
    const int N = FC1N * FC1K;   // dominant region
    prep_kernel<<<(N + 255) / 256, 256, 0, stream>>>(
        Wih1, Whh1, bih1, bhh1, Wih2, Whh2, bih2, bhh2,
        Wfc1, bfc1, Wfc2, bfc2,
        wc1, wc2, bs1, bs2, wf1, bf1, wf2, bf2);
  }

  lstm2_fused<<<Bsz / 64, 256, 0, stream>>>(X, wc1, wc2, bs1, bs2, h2g);

  {
    dim3 grid(Bsz / 128, FC1N / 64);
    fc1_gemm<<<grid, 256, 0, stream>>>(h2g, wf1, bf1, fc1o);
  }

  fc2_gemm<<<Bsz / 128, 256, 0, stream>>>(fc1o, wf2, bf2, (float*)d_out);
}
